// Decoder_60507499266374
// MI455X (gfx1250) — compile-verified, matrix operand order checked by reading, not run on hardware
//
#include <hip/hip_runtime.h>
#include <hip/hip_bf16.h>
#include <math.h>

// Problem constants (match reference)
#define E_  512
#define B_  32
#define S_  128
#define T_  128

// gfx1250 async global->LDS path (ASYNCcnt), with safe fallback
#if defined(__has_builtin)
#  if __has_builtin(__builtin_amdgcn_global_load_async_to_lds_b128) && \
      __has_builtin(__builtin_amdgcn_s_wait_asynccnt)
#    define HAVE_ASYNC_LDS 1
#  endif
#endif

typedef __attribute__((ext_vector_type(16))) __bf16 v16bf;
typedef __attribute__((ext_vector_type(8)))  float  v8f;

union FragBF { v16bf v; unsigned u[8]; };

__device__ __forceinline__ unsigned short f32_to_bf16(float f) {
  unsigned u = __float_as_uint(f);
  u += 0x7FFFu + ((u >> 16) & 1u);   // round-to-nearest-even
  return (unsigned short)(u >> 16);
}

__device__ __forceinline__ float sigm(float x) { return 1.f / (1.f + __expf(-x)); }

// Grid-wide barrier for persistent kernels (monotonic counter, zeroed per call).
__device__ __forceinline__ void grid_sync(unsigned* bar, unsigned target) {
  __threadfence();
  __syncthreads();
  if (threadIdx.x == 0) {
    atomicAdd(bar, 1u);
    while (__hip_atomic_load(bar, __ATOMIC_ACQUIRE, __HIP_MEMORY_SCOPE_AGENT) < target)
      __builtin_amdgcn_s_sleep(1);
  }
  __syncthreads();
  __builtin_amdgcn_fence(__ATOMIC_ACQUIRE, "agent");
}

// Stage `nshorts` bf16 values from global to LDS cooperatively (256 threads).
// Uses GLOBAL_LOAD_ASYNC_TO_LDS_B128 when the toolchain exposes it.
__device__ __forceinline__ void stage_h_to_lds(const unsigned short* src,
                                               unsigned short* dstLds,
                                               int nshorts, int tid) {
  const int nchunks = nshorts / 8;   // 16-byte chunks, multiple of 256
#if defined(HAVE_ASYNC_LDS)
  typedef __attribute__((ext_vector_type(4))) int vec4i_;
  typedef __attribute__((address_space(1))) vec4i_* gv4p;
  typedef __attribute__((address_space(3))) vec4i_* lv4p;
  gv4p g = (gv4p)(const void*)src;
  lv4p l = (lv4p)dstLds;
  for (int i = tid; i < nchunks; i += 256)
    __builtin_amdgcn_global_load_async_to_lds_b128(g + i, l + i, 0, 0);
  __builtin_amdgcn_s_wait_asynccnt(0);
#else
  const uint4* s4 = (const uint4*)src;
  uint4* d4 = (uint4*)dstLds;
  for (int i = tid; i < nchunks; i += 256) d4[i] = s4[i];
#endif
}

// ---------------- utility kernels ----------------

__global__ void zero_u32(unsigned* p, int n) {
  int i = blockIdx.x * blockDim.x + threadIdx.x;
  if (i < n) p[i] = 0u;
}

__global__ void cvt_f32_bf16(const float* __restrict__ in, unsigned short* __restrict__ out, int n) {
  int i = blockIdx.x * blockDim.x + threadIdx.x;
  if (i < n) out[i] = f32_to_bf16(in[i]);
}

// enc X[b*S+s][e] = relu(enc_emb[tokens[b][s]][e])  (bf16)
__global__ void enc_embed(const int* __restrict__ tokens, const float* __restrict__ emb,
                          unsigned short* __restrict__ X) {
  int i = blockIdx.x * blockDim.x + threadIdx.x;
  if (i >= B_ * S_ * E_) return;
  int e = i % E_, bs = i / E_;
  int tok = tokens[bs];
  float v = emb[(size_t)tok * E_ + e];
  X[i] = f32_to_bf16(v > 0.f ? v : 0.f);
}

// dec Xd[(b*T+t)][0..2E) = relu(dec_emb[dec_in[b][t]])  (bf16, row stride 4E)
__global__ void dec_embed(const int* __restrict__ trg, const float* __restrict__ emb,
                          unsigned short* __restrict__ Xd) {
  int i = blockIdx.x * blockDim.x + threadIdx.x;
  if (i >= B_ * T_ * 2 * E_) return;
  int j = i % (2 * E_);
  int bt = i / (2 * E_);
  int t = bt % T_, b = bt / T_;
  int tok = (t == 0) ? 1 : trg[b * T_ + t - 1];   // SOS = 1
  float v = emb[(size_t)tok * (2 * E_) + j];
  Xd[(size_t)bt * (4 * E_) + j] = f32_to_bf16(v > 0.f ? v : 0.f);
}

// dec Xd[(b*T+t)][2E..4E) = enc_hidden[b] = concat(hF[b], hB[b])
__global__ void dec_ctx_fill(const float* __restrict__ hF, const float* __restrict__ hB,
                             unsigned short* __restrict__ Xd) {
  int i = blockIdx.x * blockDim.x + threadIdx.x;
  if (i >= B_ * T_ * 2 * E_) return;
  int j = i % (2 * E_);
  int bt = i / (2 * E_);
  int b = bt / T_;
  float v = (j < E_) ? hF[b * E_ + j] : hB[b * E_ + (j - E_)];
  Xd[(size_t)bt * (4 * E_) + 2 * E_ + j] = f32_to_bf16(v);
}

// decoder h0 = enc_hidden (bf16) into ping buffer
__global__ void dec_init_state(const float* __restrict__ hF, const float* __restrict__ hB,
                               unsigned short* __restrict__ hD) {
  int i = blockIdx.x * blockDim.x + threadIdx.x;
  if (i >= B_ * 2 * E_) return;
  int j = i % (2 * E_), b = i / (2 * E_);
  float v = (j < E_) ? hF[b * E_ + j] : hB[b * E_ + (j - E_)];
  hD[i] = f32_to_bf16(v);
}

// ---------------- batched input-projection GEMM ----------------
// Fragment layouts per CDNA5 ISA 7.12.2 (wave32):
//  A 16x32 bf16: lane L holds row M=L%16; dword i holds K-pair
//     k = (i<4 ? 2i : 2i+8) + (L>=16 ? 8 : 0)
//  B 32x16 bf16: lane L holds col N=L%16; same K-pair pattern; B[k][n] = W[n][k].
//  C/D f32: element r -> M = r + (L>=16 ? 8 : 0), N = L%16.
//
// C[M x N] = A[M x K] @ W[N x K]^T + bias[n]
// 2x2 register blocking per wave: block tile 64(M) x 128(N), 8 waves.
// grid = (N/128, M/64), block = 256.
__global__ void gemm_xproj(const unsigned short* __restrict__ A,
                           const unsigned short* __restrict__ W,
                           const float* __restrict__ bias,
                           float* __restrict__ C, int N, int K) {
  const int tid  = threadIdx.x;
  const int wave = tid >> 5, lane = tid & 31;
  const int mt = wave & 1, nt = wave >> 1;
  const int m0 = blockIdx.y * 64 + mt * 32;
  const int n0 = blockIdx.x * 128 + nt * 32;
  const int l15   = lane & 15;
  const int laneK = (lane >> 4) << 3;            // 0 or 8
  const size_t rowA0 = (size_t)(m0 + l15) * K;
  const size_t rowA1 = (size_t)(m0 + 16 + l15) * K;
  const size_t rowW0 = (size_t)(n0 + l15) * K;
  const size_t rowW1 = (size_t)(n0 + 16 + l15) * K;

  v8f acc00 = {}, acc01 = {}, acc10 = {}, acc11 = {};
  for (int kk = 0; kk < K; kk += 32) {
    FragBF a0, a1, b0, b1;
#pragma unroll
    for (int i = 0; i < 8; ++i) {
      int ko = kk + laneK + (i < 4 ? 2 * i : 2 * i + 8);
      a0.u[i] = *(const unsigned*)(A + rowA0 + ko);
      a1.u[i] = *(const unsigned*)(A + rowA1 + ko);
      b0.u[i] = *(const unsigned*)(W + rowW0 + ko);
      b1.u[i] = *(const unsigned*)(W + rowW1 + ko);
    }
    acc00 = __builtin_amdgcn_wmma_f32_16x16x32_bf16(false, a0.v, false, b0.v, (short)0, acc00, false, false);
    acc01 = __builtin_amdgcn_wmma_f32_16x16x32_bf16(false, a0.v, false, b1.v, (short)0, acc01, false, false);
    acc10 = __builtin_amdgcn_wmma_f32_16x16x32_bf16(false, a1.v, false, b0.v, (short)0, acc10, false, false);
    acc11 = __builtin_amdgcn_wmma_f32_16x16x32_bf16(false, a1.v, false, b1.v, (short)0, acc11, false, false);
  }

  const int hi8 = (lane >> 4) << 3;
  const int nA = n0 + l15, nB = n0 + 16 + l15;
  const float bsA = bias[nA], bsB = bias[nB];
#pragma unroll
  for (int r = 0; r < 8; ++r) {
    int mA = m0 + r + hi8, mB = m0 + 16 + r + hi8;
    C[(size_t)mA * N + nA] = acc00[r] + bsA;
    C[(size_t)mA * N + nB] = acc01[r] + bsB;
    C[(size_t)mB * N + nA] = acc10[r] + bsA;
    C[(size_t)mB * N + nB] = acc11[r] + bsB;
  }
}

// ---------------- persistent recurrent kernels ----------------
// Block owns j-tile [jb*16, jb*16+16): computes gate columns {g*H + j} for all
// 4 gates (wave = mt*4+gate -> one 16x16 WMMA tile each), exchanges tiles via
// LDS, does the sigmoid/tanh pointwise locally with c held in registers, and
// writes the new h (bf16) to the ping-pong global buffer. One grid barrier per
// timestep. h for the step is staged into LDS (async-to-LDS when available).

// Encoder: 64 blocks = 2 dirs x 32 j-tiles. H = K = 512.
__global__ void enc_persistent(const unsigned short* __restrict__ WF,
                               const unsigned short* __restrict__ WB,
                               const float* __restrict__ GinF,
                               const float* __restrict__ GinB,
                               unsigned short* __restrict__ bufP,   // ping [dir][32][E]
                               unsigned short* __restrict__ bufQ,   // pong
                               float* __restrict__ hOutF, float* __restrict__ hOutB,
                               unsigned* __restrict__ bar) {
  const int H = E_, K = E_, NG = 4 * E_;
  extern __shared__ char smem[];
  unsigned short* hLds = (unsigned short*)smem;                 // 32*K bf16 = 32KB
  float* gLds = (float*)(smem + (size_t)32 * K * 2);            // 4*32*16 f32 = 8KB

  const int tid  = threadIdx.x;
  const int wave = tid >> 5, lane = tid & 31;
  const int mt = wave >> 2, gate = wave & 3;
  const int m_base = mt * 16;
  const int dir = blockIdx.x >> 5;
  const int jb  = blockIdx.x & 31;
  const int l15 = lane & 15;
  const int hi8 = (lane >> 4) << 3;

  const unsigned short* W = dir ? WB : WF;
  const float* Gin = dir ? GinB : GinF;
  float* hOut = dir ? hOutB : hOutF;
  const size_t dbase = (size_t)dir * 32 * K;
  const int n = gate * H + jb * 16 + l15;
  const size_t rowW = (size_t)n * K;

  unsigned short* bufOld = bufP;
  unsigned short* bufNew = bufQ;
  float creg[2] = {0.f, 0.f};

  for (int t = 0; t < S_; ++t) {
    const int tIdx = dir ? (S_ - 1 - t) : t;

    stage_h_to_lds(bufOld + dbase, hLds, 32 * K, tid);
    __syncthreads();

    // wave GEMM: 16x16 tile of h @ W^T, A from LDS, B from global (L2-hot)
    v8f acc = {};
    for (int kk = 0; kk < K; kk += 32) {
      FragBF a, b;
#pragma unroll
      for (int i = 0; i < 8; ++i) {
        int ko = kk + hi8 + (i < 4 ? 2 * i : 2 * i + 8);
        a.u[i] = *(const unsigned*)(hLds + (m_base + l15) * K + ko);
        b.u[i] = *(const unsigned*)(W + rowW + ko);
      }
      acc = __builtin_amdgcn_wmma_f32_16x16x32_bf16(false, a.v, false, b.v, (short)0, acc, false, false);
    }
#pragma unroll
    for (int r = 0; r < 8; ++r) {
      int m = m_base + r + hi8;
      gLds[(gate * 32 + m) * 16 + l15] = acc[r];
    }
    __syncthreads();

    // block-local pointwise, c in registers (2 elems/thread)
#pragma unroll
    for (int p = 0; p < 2; ++p) {
      int idx = p * 256 + tid;
      int m = idx >> 4, jl = idx & 15;
      int j = jb * 16 + jl;
      const float* grow = Gin + ((size_t)m * S_ + tIdx) * NG;
      float gi = gLds[(0 * 32 + m) * 16 + jl] + grow[0 * H + j];
      float gf = gLds[(1 * 32 + m) * 16 + jl] + grow[1 * H + j];
      float gg = gLds[(2 * 32 + m) * 16 + jl] + grow[2 * H + j];
      float go = gLds[(3 * 32 + m) * 16 + jl] + grow[3 * H + j];
      float cn = sigm(gf) * creg[p] + sigm(gi) * tanhf(gg);
      float hn = sigm(go) * tanhf(cn);
      creg[p] = cn;
      bufNew[dbase + (size_t)m * H + j] = f32_to_bf16(hn);
      if (t == S_ - 1) hOut[(size_t)m * H + j] = hn;
    }

    grid_sync(bar, 64u * (unsigned)(t + 1));
    unsigned short* tmp = bufOld; bufOld = bufNew; bufNew = tmp;
  }
}

// Decoder: 64 blocks = 64 j-tiles. H = K = 1024, writes h straight to d_out.
__global__ void dec_persistent(const unsigned short* __restrict__ W,    // 4096 x 1024
                               const float* __restrict__ Gin,           // (B*T) x 4096
                               unsigned short* __restrict__ bufP,
                               unsigned short* __restrict__ bufQ,
                               float* __restrict__ out,                 // (B,T,2E)
                               unsigned* __restrict__ bar) {
  const int H = 2 * E_, K = 2 * E_, NG = 8 * E_;
  extern __shared__ char smem[];
  unsigned short* hLds = (unsigned short*)smem;                 // 32*K bf16 = 64KB
  float* gLds = (float*)(smem + (size_t)32 * K * 2);            // 8KB

  const int tid  = threadIdx.x;
  const int wave = tid >> 5, lane = tid & 31;
  const int mt = wave >> 2, gate = wave & 3;
  const int m_base = mt * 16;
  const int jb  = blockIdx.x;              // 0..63
  const int l15 = lane & 15;
  const int hi8 = (lane >> 4) << 3;
  const int n = gate * H + jb * 16 + l15;
  const size_t rowW = (size_t)n * K;

  unsigned short* bufOld = bufP;
  unsigned short* bufNew = bufQ;
  float creg[2] = {0.f, 0.f};

  for (int t = 0; t < T_; ++t) {
    stage_h_to_lds(bufOld, hLds, 32 * K, tid);
    __syncthreads();

    v8f acc = {};
    for (int kk = 0; kk < K; kk += 32) {
      FragBF a, b;
#pragma unroll
      for (int i = 0; i < 8; ++i) {
        int ko = kk + hi8 + (i < 4 ? 2 * i : 2 * i + 8);
        a.u[i] = *(const unsigned*)(hLds + (m_base + l15) * K + ko);
        b.u[i] = *(const unsigned*)(W + rowW + ko);
      }
      acc = __builtin_amdgcn_wmma_f32_16x16x32_bf16(false, a.v, false, b.v, (short)0, acc, false, false);
    }
#pragma unroll
    for (int r = 0; r < 8; ++r) {
      int m = m_base + r + hi8;
      gLds[(gate * 32 + m) * 16 + l15] = acc[r];
    }
    __syncthreads();

#pragma unroll
    for (int p = 0; p < 2; ++p) {
      int idx = p * 256 + tid;
      int m = idx >> 4, jl = idx & 15;
      int j = jb * 16 + jl;
      const float* grow = Gin + ((size_t)m * T_ + t) * NG;
      float gi = gLds[(0 * 32 + m) * 16 + jl] + grow[0 * H + j];
      float gf = gLds[(1 * 32 + m) * 16 + jl] + grow[1 * H + j];
      float gg = gLds[(2 * 32 + m) * 16 + jl] + grow[2 * H + j];
      float go = gLds[(3 * 32 + m) * 16 + jl] + grow[3 * H + j];
      float cn = sigm(gf) * creg[p] + sigm(gi) * tanhf(gg);
      float hn = sigm(go) * tanhf(cn);
      creg[p] = cn;
      bufNew[(size_t)m * H + j] = f32_to_bf16(hn);
      out[((size_t)m * T_ + t) * H + j] = hn;
    }

    grid_sync(bar, 64u * (unsigned)(t + 1));
    unsigned short* tmp = bufOld; bufOld = bufNew; bufNew = tmp;
  }
}

// ---------------- launcher ----------------

extern "C" void kernel_launch(void* const* d_in, const int* in_sizes, int n_in,
                              void* d_out, int out_size, void* d_ws, size_t ws_size,
                              hipStream_t stream) {
  (void)in_sizes; (void)n_in; (void)out_size; (void)ws_size;

  const int*   tokens  = (const int*)d_in[0];
  const int*   trg     = (const int*)d_in[1];
  const float* enc_emb = (const float*)d_in[2];
  const float* dec_emb = (const float*)d_in[3];
  const float* eWihF   = (const float*)d_in[4];
  const float* eWhhF   = (const float*)d_in[5];
  const float* ebF     = (const float*)d_in[6];
  const float* eWihB   = (const float*)d_in[7];
  const float* eWhhB   = (const float*)d_in[8];
  const float* ebB     = (const float*)d_in[9];
  const float* dWih    = (const float*)d_in[10];
  const float* dWhh    = (const float*)d_in[11];
  const float* db      = (const float*)d_in[12];
  float* out = (float*)d_out;

  char* ws = (char*)d_ws;
  size_t off = 0;
  auto alloc = [&](size_t bytes) -> char* {
    char* p = ws + off;
    off = (off + bytes + 255) & ~(size_t)255;
    return p;
  };

  const int EW  = 4 * E_ * E_;           // 2048x512
  const int DWI = 8 * E_ * 4 * E_;       // 4096x2048
  const int DWH = 8 * E_ * 2 * E_;       // 4096x1024
  unsigned short* wEWihF = (unsigned short*)alloc((size_t)EW * 2);
  unsigned short* wEWhhF = (unsigned short*)alloc((size_t)EW * 2);
  unsigned short* wEWihB = (unsigned short*)alloc((size_t)EW * 2);
  unsigned short* wEWhhB = (unsigned short*)alloc((size_t)EW * 2);
  unsigned short* wDWih  = (unsigned short*)alloc((size_t)DWI * 2);
  unsigned short* wDWhh  = (unsigned short*)alloc((size_t)DWH * 2);

  unsigned short* Xe = (unsigned short*)alloc((size_t)B_ * S_ * E_ * 2);
  unsigned short* Xd = (unsigned short*)alloc((size_t)B_ * T_ * 4 * E_ * 2);

  float* GinF = (float*)alloc((size_t)B_ * S_ * 4 * E_ * 4);
  float* GinB = (float*)alloc((size_t)B_ * S_ * 4 * E_ * 4);
  float* GinD = (float*)alloc((size_t)B_ * T_ * 8 * E_ * 4);

  unsigned short* hPingE = (unsigned short*)alloc((size_t)2 * B_ * E_ * 2);
  unsigned short* hPongE = (unsigned short*)alloc((size_t)2 * B_ * E_ * 2);
  unsigned short* hPingD = (unsigned short*)alloc((size_t)B_ * 2 * E_ * 2);
  unsigned short* hPongD = (unsigned short*)alloc((size_t)B_ * 2 * E_ * 2);
  float* hOutF = (float*)alloc((size_t)B_ * E_ * 4);
  float* hOutB = (float*)alloc((size_t)B_ * E_ * 4);
  unsigned* bar = (unsigned*)alloc(256);

  const int TB = 256;
  auto g1 = [](int n) { return dim3((unsigned)((n + 255) / 256)); };

  // weight conversion + embeddings
  cvt_f32_bf16<<<g1(EW), TB, 0, stream>>>(eWihF, wEWihF, EW);
  cvt_f32_bf16<<<g1(EW), TB, 0, stream>>>(eWhhF, wEWhhF, EW);
  cvt_f32_bf16<<<g1(EW), TB, 0, stream>>>(eWihB, wEWihB, EW);
  cvt_f32_bf16<<<g1(EW), TB, 0, stream>>>(eWhhB, wEWhhB, EW);
  cvt_f32_bf16<<<g1(DWI), TB, 0, stream>>>(dWih, wDWih, DWI);
  cvt_f32_bf16<<<g1(DWH), TB, 0, stream>>>(dWhh, wDWhh, DWH);
  enc_embed<<<g1(B_ * S_ * E_), TB, 0, stream>>>(tokens, enc_emb, Xe);
  dec_embed<<<g1(B_ * T_ * 2 * E_), TB, 0, stream>>>(trg, dec_emb, Xd);

  // zero h0 (encoder ping) and barrier counters (every call: graph-replay safe)
  zero_u32<<<g1(2 * B_ * E_ / 2), TB, 0, stream>>>((unsigned*)hPingE, 2 * B_ * E_ / 2);
  zero_u32<<<dim3(1), 64, 0, stream>>>(bar, 64);

  // batched encoder input projections: M=4096, K=512, N=2048
  gemm_xproj<<<dim3((4 * E_) / 128, (B_ * S_) / 64), 256, 0, stream>>>(
      Xe, wEWihF, ebF, GinF, 4 * E_, E_);
  gemm_xproj<<<dim3((4 * E_) / 128, (B_ * S_) / 64), 256, 0, stream>>>(
      Xe, wEWihB, ebB, GinB, 4 * E_, E_);

  // persistent encoder recurrence (both directions, 128 steps, 1 launch)
  enc_persistent<<<dim3(64), 256, 32 * E_ * 2 + 4 * 32 * 16 * 4, stream>>>(
      wEWhhF, wEWhhB, GinF, GinB, hPingE, hPongE, hOutF, hOutB, bar + 0);

  // decoder inputs
  dec_ctx_fill<<<g1(B_ * T_ * 2 * E_), TB, 0, stream>>>(hOutF, hOutB, Xd);
  dec_init_state<<<g1(B_ * 2 * E_), TB, 0, stream>>>(hOutF, hOutB, hPingD);

  // batched decoder input projection: M=4096, K=2048, N=4096
  gemm_xproj<<<dim3((8 * E_) / 128, (B_ * T_) / 64), 256, 0, stream>>>(
      Xd, wDWih, db, GinD, 8 * E_, 4 * E_);

  // persistent decoder recurrence (128 steps, 1 launch), h -> d_out
  dec_persistent<<<dim3(64), 256, 32 * 2 * E_ * 2 + 4 * 32 * 16 * 4, stream>>>(
      wDWhh, GinD, hPingD, hPongD, out, bar + 1);
}
